// Attention_77292231459044
// MI455X (gfx1250) — matrix operand order repaired and checked by reading, not compiled, not run on hardware
//
#include <hip/hip_runtime.h>
#include <hip/hip_bf16.h>
#include <stdint.h>

// ---------------------------------------------------------------------------
// CDNA5 (gfx1250) wave32 WMMA attention pipeline:
//   K1: qkv  = x @ W_qkv + b        (f32 in -> bf16 out, WMMA bf16)
//   K2: flash attention per (b,h)   (TDM-staged Q/K, bf16 WMMA, f32 softmax)
//   K3: out  = y @ W_proj + b       (bf16 in -> f32 out, WMMA bf16)
// ---------------------------------------------------------------------------

typedef __bf16 bf16_t;
typedef bf16_t v16bf __attribute__((ext_vector_type(16)));
typedef float  v8f   __attribute__((ext_vector_type(8)));
typedef unsigned int v4u __attribute__((ext_vector_type(4)));
typedef int v8i __attribute__((ext_vector_type(8)));
typedef int v4i __attribute__((ext_vector_type(4)));

struct Frag { uint32_t u[8]; };   // 16 packed bf16 = one WMMA A/B operand

__device__ __forceinline__ v8f wmma_bf16(const Frag& a, const Frag& b, v8f c) {
    return __builtin_amdgcn_wmma_f32_16x16x32_bf16(
        /*neg_a=*/false, __builtin_bit_cast(v16bf, a),
        /*neg_b=*/false, __builtin_bit_cast(v16bf, b),
        /*c_mod=*/(short)0, c, /*reuse_a=*/false, /*reuse_b=*/false);
}

// f32 -> bf16 (round to nearest even), branch-free.
__device__ __forceinline__ unsigned short f2bf(float f) {
    uint32_t u = __builtin_bit_cast(uint32_t, f);
    u += 0x7fffu + ((u >> 16) & 1u);
    return (unsigned short)(u >> 16);
}

// A-fragment (16x32 bf16): caller passes &row[kh] where kh = (lane<16)?0:8.
__device__ __forceinline__ void load_a16(Frag& f, const unsigned short* p) {
    uint4 x0 = *reinterpret_cast<const uint4*>(p);
    uint4 x1 = *reinterpret_cast<const uint4*>(p + 16);
    f.u[0]=x0.x; f.u[1]=x0.y; f.u[2]=x0.z; f.u[3]=x0.w;
    f.u[4]=x1.x; f.u[5]=x1.y; f.u[6]=x1.z; f.u[7]=x1.w;
}

// B-fragment (32x16 bf16), LDS stored [N][K]: caller passes &col[kb],
// kb = (lane<16)?0:8; K values kb..kb+7 and kb+16..kb+23 per lane.
__device__ __forceinline__ void load_b16(Frag& f, const unsigned short* p) {
    uint4 x0 = *reinterpret_cast<const uint4*>(p);
    uint4 x1 = *reinterpret_cast<const uint4*>(p + 16);
    f.u[0]=x0.x; f.u[1]=x0.y; f.u[2]=x0.z; f.u[3]=x0.w;
    f.u[4]=x1.x; f.u[5]=x1.y; f.u[6]=x1.z; f.u[7]=x1.w;
}

// ---------------------------------------------------------------------------
// Tensor Data Mover: 2D tile (bf16) global -> LDS.  ISA 08_async_tensor §8.
// D# group0: count=1 | lds_addr | global_addr[56:0] | type=2.
// D# group1: data_size=2B, tensor_dim0/1 == tile_dim0/1, dim0 stride.
// Issued by one wave; completion via s_wait_tensorcnt 0, published by barrier.
// ---------------------------------------------------------------------------
__device__ __forceinline__ void tdm_load_2d_bf16(const void* lds_dst,
                                                 const void* global_src,
                                                 uint32_t width_elems,
                                                 uint32_t rows,
                                                 uint32_t row_stride_elems) {
    const uint64_t ga  = (uint64_t)(uintptr_t)global_src;
    const uint32_t lds = (uint32_t)(uintptr_t)lds_dst;   // low 32 bits = LDS offset

    union { uint32_t w[4]; v4u v; } g0 = {};
    g0.w[0] = 1u;                                    // count=1, user descriptor
    g0.w[1] = lds;                                   // lds_addr  [63:32]
    g0.w[2] = (uint32_t)ga;                          // global_addr[95:64]
    g0.w[3] = (uint32_t)(ga >> 32) | (2u << 30);     // addr[56:32], type=2

    union { uint32_t w[8]; v8i v; } g1 = {};
    g1.w[0] = (1u << 16);                            // data_size = 2 bytes
    g1.w[1] = (width_elems & 0xffffu) << 16;         // tensor_dim0 lo16 @ [63:48]
    g1.w[2] = (width_elems >> 16) |                  // tensor_dim0 hi16
              ((rows & 0xffffu) << 16);              // tensor_dim1 lo16
    g1.w[3] = (rows >> 16) |                         // tensor_dim1 hi16
              (width_elems << 16);                   // tile_dim0  @ [127:112]
    g1.w[4] = rows & 0xffffu;                        // tile_dim1 (tile_dim2 = 0)
    g1.w[5] = row_stride_elems;                      // tensor_dim0_stride lo32
    g1.w[6] = 0;                                     // stride hi16 / dim1_stride
    g1.w[7] = 0;

    const v4i z4 = {0, 0, 0, 0};
#if __clang_major__ >= 23
    const v8i z8 = {0, 0, 0, 0, 0, 0, 0, 0};
    __builtin_amdgcn_tensor_load_to_lds(g0.v, g1.v, z4, z4, z8, 0);
#else
    __builtin_amdgcn_tensor_load_to_lds(g0.v, g1.v, z4, z4, 0);
#endif
}

// ---------------------------------------------------------------------------
// GEMM: out[M,N] = A[M,K] * W[K,N] + bias,  bf16 WMMA, f32 accumulate.
// Block tile 128x64, BK=32, 8 waves (256 thr) each own a 32x32 macro-tile.
// ---------------------------------------------------------------------------
#define GBM 128
#define GBN 64
#define GBK 32

template <bool A_IS_F32, bool OUT_BF16>
__global__ __launch_bounds__(256)
void gemm_wmma_bf16(const void* __restrict__ Ap, const float* __restrict__ W,
                    const float* __restrict__ bias, void* __restrict__ outp,
                    int M, int N, int K) {
    __shared__ unsigned short sA[2][GBM * GBK];   // [m][k]  16 KB
    __shared__ unsigned short sB[2][GBN * GBK];   // [n][k]   8 KB

    const int tid  = threadIdx.x;
    const int lane = tid & 31;
    const int wave = tid >> 5;
    const int wm   = wave & 3;       // 4 waves along M
    const int wn   = wave >> 2;      // 2 waves along N
    const int m0   = blockIdx.y * GBM;
    const int n0   = blockIdx.x * GBN;

    auto stageA = [&](int buf, int k0) {
        if constexpr (A_IS_F32) {
            const float* A = (const float*)Ap;
#pragma unroll
            for (int i = 0; i < 4; ++i) {
                int e = (tid + i * 256) * 4;          // 4096 f32 elements
                int r = e >> 5, c = e & 31;
                float4 f = *reinterpret_cast<const float4*>(
                    &A[(size_t)(m0 + r) * K + k0 + c]);
                unsigned short* d = &sA[buf][r * GBK + c];
                d[0] = f2bf(f.x); d[1] = f2bf(f.y);
                d[2] = f2bf(f.z); d[3] = f2bf(f.w);
            }
        } else {
            const unsigned short* A = (const unsigned short*)Ap;
#pragma unroll
            for (int i = 0; i < 2; ++i) {
                int e = (tid + i * 256) * 8;          // 4096 bf16 elements
                int r = e >> 5, c = e & 31;
                uint4 v = *reinterpret_cast<const uint4*>(
                    &A[(size_t)(m0 + r) * K + k0 + c]);
                *reinterpret_cast<uint4*>(&sA[buf][r * GBK + c]) = v;
            }
        }
    };
    auto stageB = [&](int buf, int k0) {
#pragma unroll
        for (int i = 0; i < 8; ++i) {
            int e  = tid + i * 256;                   // 2048 elements
            int kk = e >> 6, nn = e & 63;
            float w = W[(size_t)(k0 + kk) * N + n0 + nn];
            sB[buf][nn * GBK + kk] = f2bf(w);         // transpose into [n][k]
        }
    };

    v8f c00 = {0,0,0,0,0,0,0,0}, c01 = {0,0,0,0,0,0,0,0};
    v8f c10 = {0,0,0,0,0,0,0,0}, c11 = {0,0,0,0,0,0,0,0};

    stageA(0, 0);
    stageB(0, 0);

    const int rA = lane & 15;
    const int kh = (lane < 16) ? 0 : 8;    // A K-half select
    const int kb = (lane < 16) ? 0 : 8;    // B K-half select
    const int nk = K / GBK;

    for (int kt = 0; kt < nk; ++kt) {
        __syncthreads();
        const int buf = kt & 1;
        if (kt + 1 < nk) {
            stageA(buf ^ 1, (kt + 1) * GBK);
            stageB(buf ^ 1, (kt + 1) * GBK);
        }
        if (kt + 2 < nk) {   // global_prefetch_b8 for the K-tile two ahead
            __builtin_prefetch(&W[(size_t)((kt + 2) * GBK) * N + n0 + (tid & 63)], 0, 1);
        }
        const unsigned short* As = sA[buf];
        const unsigned short* Bs = sB[buf];

        Frag a0, a1, b0, b1;
        load_a16(a0, &As[(wm * 32 +      rA) * GBK + kh]);
        load_a16(a1, &As[(wm * 32 + 16 + rA) * GBK + kh]);
        load_b16(b0, &Bs[(wn * 32 +      rA) * GBK + kb]);
        load_b16(b1, &Bs[(wn * 32 + 16 + rA) * GBK + kb]);

        c00 = wmma_bf16(a0, b0, c00);
        c01 = wmma_bf16(a0, b1, c01);
        c10 = wmma_bf16(a1, b0, c10);
        c11 = wmma_bf16(a1, b1, c11);
    }

    // Epilogue: C layout (ISA 7.12.2): row = r0+i, col = lane&15.
    const int col = lane & 15;
    const int r0  = (lane < 16) ? 0 : 8;
    auto storeTile = [&](const v8f& c, int ti, int tj) {
        int gn = n0 + wn * 32 + tj * 16 + col;
        float bv = bias[gn];
        int gm = m0 + wm * 32 + ti * 16 + r0;
#pragma unroll
        for (int i = 0; i < 8; ++i) {
            float v = c[i] + bv;
            size_t off = (size_t)(gm + i) * N + gn;
            if constexpr (OUT_BF16) ((unsigned short*)outp)[off] = f2bf(v);
            else                    ((float*)outp)[off] = v;
        }
    };
    storeTile(c00, 0, 0); storeTile(c01, 0, 1);
    storeTile(c10, 1, 0); storeTile(c11, 1, 1);
}

// ---------------------------------------------------------------------------
// Flash attention (causal).  qkv: bf16 [B*T][3072]  (q|k|v, 16 heads x 64).
// Grid: (T/128, B*H). Block: 256 thr = 8 waves, 16 query rows per wave.
// Q and K tiles are staged by the Tensor Data Mover (wave 0 issues the D#,
// waits on TENSORcnt, workgroup barrier publishes the LDS tile).
// ---------------------------------------------------------------------------
#define FBQ 128
#define FKV 32
#define FHD 64

__global__ __launch_bounds__(256)
void flash_attn_wmma(const unsigned short* __restrict__ qkv,
                     unsigned short* __restrict__ y, int T) {
    __shared__ unsigned short sQ[FBQ * FHD];       // [q][d]   16 KB
    __shared__ unsigned short sK[FKV * FHD];       // [key][d]  4 KB
    __shared__ unsigned short sV[FHD * FKV];       // [d][key]  4 KB (transposed)
    __shared__ unsigned short sP[8][16 * FKV];     // per-wave P 8 KB

    const int tid  = threadIdx.x;
    const int lane = tid & 31;
    const int wave = tid >> 5;
    const int q0   = blockIdx.x * FBQ;
    const int bh   = blockIdx.y;
    const int b    = bh >> 4, h = bh & 15;
    const size_t rowBase = (size_t)b * T;
    const int LDQ = 3072;
    const int qcol = h * FHD, kcol = 1024 + h * FHD, vcol = 2048 + h * FHD;

    // ---- stage Q tile (128 rows x 64 bf16, row stride 3072) via TDM ----
    if (wave == 0) {
        tdm_load_2d_bf16(sQ, &qkv[(rowBase + q0) * LDQ + qcol],
                         FHD, FBQ, LDQ);
        __builtin_amdgcn_s_wait_tensorcnt(0);
    }
    __syncthreads();

    const int rA   = lane & 15;
    const int kh   = (lane < 16) ? 0 : 8;
    const int kb   = (lane < 16) ? 0 : 8;
    const int colK = lane & 15;
    const int rowOff = (lane < 16) ? 0 : 8;        // C-layout row offset

    Frag aq0, aq1;                                  // Q 16x64: two K=32 slices
    load_a16(aq0, &sQ[(wave * 16 + rA) * FHD +  0 + kh]);
    load_a16(aq1, &sQ[(wave * 16 + rA) * FHD + 32 + kh]);

    v8f o0 = {0,0,0,0,0,0,0,0}, o1 = {0,0,0,0,0,0,0,0};
    v8f o2 = {0,0,0,0,0,0,0,0}, o3 = {0,0,0,0,0,0,0,0};
    float mrow[8], lrow[8];
#pragma unroll
    for (int i = 0; i < 8; ++i) { mrow[i] = -__builtin_inff(); lrow[i] = 0.f; }

    const float SCL = 0.125f * 1.44269504088896340736f;  // 1/sqrt(64) * log2(e)
    const int qEnd = q0 + FBQ - 1;

    for (int k0 = 0; k0 <= qEnd; k0 += FKV) {
        // ---- stage K (32x64, [key][d]) via TDM; V transposed by VALU ----
        if (wave == 0) {
            tdm_load_2d_bf16(sK, &qkv[(rowBase + k0) * LDQ + kcol],
                             FHD, FKV, LDQ);
        }
        {
            int e = tid * 8;
            int key = e >> 6, d = e & 63;
            uint4 vv = *reinterpret_cast<const uint4*>(
                &qkv[(rowBase + k0 + key) * LDQ + vcol + d]);
            uint32_t w0 = vv.x, w1 = vv.y, w2 = vv.z, w3 = vv.w;
            sV[(d + 0) * FKV + key] = (unsigned short)(w0 & 0xffffu);
            sV[(d + 1) * FKV + key] = (unsigned short)(w0 >> 16);
            sV[(d + 2) * FKV + key] = (unsigned short)(w1 & 0xffffu);
            sV[(d + 3) * FKV + key] = (unsigned short)(w1 >> 16);
            sV[(d + 4) * FKV + key] = (unsigned short)(w2 & 0xffffu);
            sV[(d + 5) * FKV + key] = (unsigned short)(w2 >> 16);
            sV[(d + 6) * FKV + key] = (unsigned short)(w3 & 0xffffu);
            sV[(d + 7) * FKV + key] = (unsigned short)(w3 >> 16);
        }
        if (wave == 0) {
            __builtin_amdgcn_s_wait_tensorcnt(0);
        }
        __syncthreads();

        // ---- S = Q K^T : two 16x16 key tiles, each over two d-slices ----
        Frag b00, b01, b10, b11;
        load_b16(b00, &sK[(     colK) * FHD +  0 + kb]);   // keys 0..15, d 0..31
        load_b16(b01, &sK[(     colK) * FHD + 32 + kb]);   // keys 0..15, d 32..63
        load_b16(b10, &sK[(16 + colK) * FHD +  0 + kb]);   // keys 16..31
        load_b16(b11, &sK[(16 + colK) * FHD + 32 + kb]);

        v8f s0 = {0,0,0,0,0,0,0,0}, s1 = {0,0,0,0,0,0,0,0};
        s0 = wmma_bf16(aq0, b00, s0);  s0 = wmma_bf16(aq1, b01, s0);
        s1 = wmma_bf16(aq0, b10, s1);  s1 = wmma_bf16(aq1, b11, s1);

        // ---- online softmax (exp2 space) ----
        unsigned short* Pw = sP[wave];
#pragma unroll
        for (int i = 0; i < 8; ++i) {
            int qr  = q0 + wave * 16 + rowOff + i;
            int kc0 = k0 + colK, kc1 = k0 + 16 + colK;
            float v0 = (kc0 <= qr) ? s0[i] * SCL : -__builtin_inff();
            float v1 = (kc1 <= qr) ? s1[i] * SCL : -__builtin_inff();
            float mx = fmaxf(v0, v1);
            mx = fmaxf(mx, __shfl_xor(mx, 1));
            mx = fmaxf(mx, __shfl_xor(mx, 2));
            mx = fmaxf(mx, __shfl_xor(mx, 4));
            mx = fmaxf(mx, __shfl_xor(mx, 8));
            float mold = mrow[i];
            float mnew = fmaxf(mold, mx);
            bool  inval = (mnew == -__builtin_inff());
            float alpha = inval ? 1.f : exp2f(mold - mnew);
            float p0 = inval ? 0.f : exp2f(v0 - mnew);
            float p1 = inval ? 0.f : exp2f(v1 - mnew);
            float rs = p0 + p1;
            rs += __shfl_xor(rs, 1);
            rs += __shfl_xor(rs, 2);
            rs += __shfl_xor(rs, 4);
            rs += __shfl_xor(rs, 8);
            lrow[i] = lrow[i] * alpha + rs;
            mrow[i] = mnew;
            o0[i] *= alpha; o1[i] *= alpha; o2[i] *= alpha; o3[i] *= alpha;
            // C-layout -> A-layout via per-wave LDS (in-order per wave)
            Pw[(rowOff + i) * FKV +      colK] = f2bf(p0);
            Pw[(rowOff + i) * FKV + 16 + colK] = f2bf(p1);
        }

        // ---- O += P V ----
        Frag ap;
        load_a16(ap, &Pw[rA * FKV + kh]);
        Frag bv0, bv1, bv2, bv3;
        load_b16(bv0, &sV[( 0 + colK) * FKV + kb]);
        load_b16(bv1, &sV[(16 + colK) * FKV + kb]);
        load_b16(bv2, &sV[(32 + colK) * FKV + kb]);
        load_b16(bv3, &sV[(48 + colK) * FKV + kb]);
        o0 = wmma_bf16(ap, bv0, o0);
        o1 = wmma_bf16(ap, bv1, o1);
        o2 = wmma_bf16(ap, bv2, o2);
        o3 = wmma_bf16(ap, bv3, o3);

        __syncthreads();   // protect sK/sV before next stage
    }

    // ---- epilogue: y[b*T+q][h*64+d] = O / l  (bf16) ----
#pragma unroll
    for (int i = 0; i < 8; ++i) {
        float inv = 1.0f / lrow[i];
        size_t row = rowBase + (size_t)(q0 + wave * 16 + rowOff + i);
        size_t base = row * 1024 + (size_t)h * FHD;
        y[base +  0 + colK] = f2bf(o0[i] * inv);
        y[base + 16 + colK] = f2bf(o1[i] * inv);
        y[base + 32 + colK] = f2bf(o2[i] * inv);
        y[base + 48 + colK] = f2bf(o3[i] * inv);
    }
}

// ---------------------------------------------------------------------------
extern "C" void kernel_launch(void* const* d_in, const int* in_sizes, int n_in,
                              void* d_out, int out_size, void* d_ws, size_t ws_size,
                              hipStream_t stream) {
    (void)in_sizes; (void)n_in; (void)out_size; (void)ws_size;
    const float* x      = (const float*)d_in[0];
    const float* W_qkv  = (const float*)d_in[1];
    const float* b_qkv  = (const float*)d_in[2];
    const float* W_proj = (const float*)d_in[3];
    const float* b_proj = (const float*)d_in[4];
    float* out = (float*)d_out;

    const int B = 4, T = 2048, C = 1024;
    const int M = B * T;                       // 8192

    unsigned short* qkv = (unsigned short*)d_ws;              // [M][3C] bf16
    unsigned short* yws = qkv + (size_t)M * 3 * C;            // [M][C]  bf16

    dim3 blk(256);
    // K1: QKV projection, bf16 output
    gemm_wmma_bf16<true, true>
        <<<dim3((3 * C) / GBN, M / GBM), blk, 0, stream>>>(
            x, W_qkv, b_qkv, qkv, M, 3 * C, C);
    // K2: causal flash attention (TDM-staged Q/K)
    flash_attn_wmma
        <<<dim3(T / FBQ, B * 16), blk, 0, stream>>>(qkv, yws, T);
    // K3: output projection, f32 output
    gemm_wmma_bf16<false, false>
        <<<dim3(C / GBN, M / GBM), blk, 0, stream>>>(
            yws, W_proj, b_proj, out, M, C, C);
}
